// GraphConv_79139067396228
// MI455X (gfx1250) — matrix operand order repaired and checked by reading, not compiled, not run on hardware
//
#include <hip/hip_runtime.h>

typedef __attribute__((ext_vector_type(16))) __bf16        v16bf;
typedef __attribute__((ext_vector_type(8)))  float         v8f;
typedef __attribute__((ext_vector_type(4)))  unsigned int  v4u;
typedef __attribute__((ext_vector_type(8)))  unsigned int  v8u;

#define NNODES 50000
#define NEDGES 600000
#define INCH   128
#define NCOL   128      // HEADS * OUT_CH
#define NHEADS 4
#define OUTCH  32
#define MT     3125     // NNODES / 16 (exact)
#define KB     4        // 128 / 32
#define NT     8        // 128 / 16
#define BELEMS (NT * KB * 32 * 16)   // 16384 ushorts = 32 KB staged B

// ---- helpers ----
__device__ __forceinline__ unsigned short f2bf(float f) {
    unsigned int u = __float_as_uint(f);
    unsigned int r = (u + 0x7FFFu + ((u >> 16) & 1u)) >> 16;   // RNE
    return (unsigned short)r;
}
__device__ __forceinline__ unsigned int mapf(float f) {        // order-preserving float->uint
    unsigned int b = __float_as_uint(f);
    return (b & 0x80000000u) ? ~b : (b | 0x80000000u);
}
__device__ __forceinline__ float unmapf(unsigned int u) {
    return __uint_as_float((u & 0x80000000u) ? (u ^ 0x80000000u) : ~u);
}

// ---- zero-init out / amax / denom ----
__global__ void k_init(float* __restrict__ out, unsigned int* __restrict__ amax,
                       float* __restrict__ denom) {
    int t = blockIdx.x * blockDim.x + threadIdx.x;
    if (t < NNODES * NCOL) out[t] = 0.f;
    if (t < NNODES * NHEADS) { amax[t] = 0u; denom[t] = 0.f; }
}

// ---- stage B (weight 128x128 row-major) into WMMA 16-bit B 32x16 lane layout ----
__global__ void k_stage_b(const float* __restrict__ W, unsigned short* __restrict__ Bst) {
    int t = blockIdx.x * blockDim.x + threadIdx.x;
    if (t >= NT * KB * 32) return;
    int lane = t & 31;
    int kb   = (t >> 5) & 3;
    int n    = (t >> 7) * 16 + (lane & 15);
    int kbase = kb * 32 + ((lane >= 16) ? 8 : 0);
    unsigned short* dst = Bst + (size_t)t * 16;
    #pragma unroll
    for (int e = 0; e < 16; ++e)
        dst[e] = f2bf(W[(size_t)(kbase + (e < 8 ? e : e + 8)) * NCOL + n]);
}

// ---- GEMM h = x @ W via v_wmma_f32_16x16x32_bf16, B staged in LDS via TDM ----
__global__ void k_gemm(const float* __restrict__ x,
                       const unsigned short* __restrict__ Bst,
                       float* __restrict__ h) {
    __shared__ unsigned short Bsh[BELEMS];   // 32 KB

    // One wave DMAs the whole staged B into LDS with the Tensor Data Mover:
    // D# describes a 1-row tile of 4096 x 8-byte elements (32 KB contiguous).
    if (threadIdx.x < 32) {
        unsigned long long ga = (unsigned long long)(const void*)Bst;
        unsigned int lds = (unsigned int)(unsigned long long)(void*)&Bsh[0]; // aperture low bits = LDS offset
        v4u g0 = { 1u,                                        // count=1, is_restore=0
                   lds,                                       // lds_addr
                   (unsigned int)(ga & 0xFFFFFFFFu),          // global_addr[31:0]
                   (unsigned int)((ga >> 32) & 0x01FFFFFFu)   // global_addr[56:32]
                     | 0x80000000u };                         // type=2 ("image")
        v8u g1 = { 0x00030000u,   // wg_mask=0, data_size=3 (8B), no iterate/pad
                   0x10000000u,   // tensor_dim0 = 4096 (lo16 in [31:16])
                   0x00010000u,   // tensor_dim0 hi16 = 0, tensor_dim1 = 1
                   0x10000000u,   // tensor_dim1 hi16 = 0, tile_dim0 = 4096
                   0x00000001u,   // tile_dim1 = 1, tile_dim2 = 0
                   4096u,         // tensor_dim0_stride[31:0]
                   0u, 0u };
        v4u gz = { 0u, 0u, 0u, 0u };
        asm volatile("tensor_load_to_lds %0, %1, %2, %3"
                     :: "s"(g0), "s"(g1), "s"(gz), "s"(gz) : "memory");
        __builtin_amdgcn_s_wait_tensorcnt((short)0);
    }
    __syncthreads();

    int lane = threadIdx.x & 31;
    int mt   = blockIdx.x * (blockDim.x >> 5) + (threadIdx.x >> 5);
    if (mt < MT) {
        v8f acc[NT];
        #pragma unroll
        for (int i = 0; i < NT; ++i) acc[i] = (v8f){0,0,0,0,0,0,0,0};

        int m  = mt * 16 + (lane & 15);
        int hi = (lane >= 16) ? 8 : 0;

        #pragma unroll
        for (int kb = 0; kb < KB; ++kb) {
            // A layout (16-bit 16x32): lane<16 row=lane, elems 0..7=K0..7, 8..15=K16..23;
            //                          lane>=16 row=lane-16, elems 0..7=K8..15, 8..15=K24..31.
            const float* xp = x + (size_t)m * INCH + kb * 32 + hi;
            v8f lo8 = *(const v8f*)(xp);
            v8f hi8 = *(const v8f*)(xp + 16);
            v16bf a;
            #pragma unroll
            for (int e = 0; e < 8; ++e) {
                a[e]     = (__bf16)lo8[e];
                a[e + 8] = (__bf16)hi8[e];
            }
            #pragma unroll
            for (int nt = 0; nt < NT; ++nt) {
                v16bf b = *(const v16bf*)(&Bsh[((nt * KB + kb) * 32 + lane) * 16]);
                acc[nt] = __builtin_amdgcn_wmma_f32_16x16x32_bf16(
                    false, a, false, b, (short)0, acc[nt], false, false);
            }
        }

        // C/D layout: lane<16 -> N=lane, M=r; lane>=16 -> N=lane-16, M=8+r
        int mrow = mt * 16 + hi;
        int nc   = lane & 15;
        #pragma unroll
        for (int nt = 0; nt < NT; ++nt)
            #pragma unroll
            for (int r = 0; r < 8; ++r)
                h[(size_t)(mrow + r) * NCOL + nt * 16 + nc] = acc[nt][r];
    }
}

// ---- per-node attention scores: s_l[n,h] = h[n,h,:].wl ; s_r = h[n,h,:].wr ----
__global__ void k_scores(const float* __restrict__ h, const float* __restrict__ att,
                         float* __restrict__ sl, float* __restrict__ sr) {
    int t = blockIdx.x * blockDim.x + threadIdx.x;
    if (t >= NNODES * NHEADS) return;
    int n = t >> 2, hd = t & 3;
    const float* hp = h + (size_t)n * NCOL + hd * OUTCH;
    const float* wl = att + hd * 2 * OUTCH;
    const float* wr = wl + OUTCH;
    float a = 0.f, b = 0.f;
    #pragma unroll
    for (int c = 0; c < OUTCH; ++c) { float v = hp[c]; a += v * wl[c]; b += v * wr[c]; }
    sl[t] = a; sr[t] = b;
}

// ---- segment max over row (edges + self loops) ----
__global__ void k_edge_max(const int* __restrict__ ei,
                           const float* __restrict__ sl, const float* __restrict__ sr,
                           unsigned int* __restrict__ amax) {
    int e = blockIdx.x * blockDim.x + threadIdx.x;
    if (e >= NEDGES + NNODES) return;
    int row, col;
    if (e < NEDGES) { row = ei[e]; col = ei[NEDGES + e]; }
    else            { row = col = e - NEDGES; }
    #pragma unroll
    for (int hd = 0; hd < NHEADS; ++hd) {
        float a = sl[row * NHEADS + hd] + sr[col * NHEADS + hd];
        a = (a > 0.f) ? a : 0.2f * a;                 // leaky relu
        atomicMax(&amax[row * NHEADS + hd], mapf(a));
    }
}

// ---- accumulate exp weights: denom += w ; out[row] += w * h[col] ----
__global__ void k_edge_agg(const int* __restrict__ ei,
                           const float* __restrict__ sl, const float* __restrict__ sr,
                           const unsigned int* __restrict__ amax,
                           const float* __restrict__ h,
                           float* __restrict__ denom, float* __restrict__ out) {
    int t = blockIdx.x * blockDim.x + threadIdx.x;
    if (t >= (NEDGES + NNODES) * NHEADS) return;
    int e = t >> 2, hd = t & 3;
    int row, col;
    if (e < NEDGES) { row = ei[e]; col = ei[NEDGES + e]; }
    else            { row = col = e - NEDGES; }
    float a = sl[row * NHEADS + hd] + sr[col * NHEADS + hd];
    a = (a > 0.f) ? a : 0.2f * a;
    float w = __expf(a - unmapf(amax[row * NHEADS + hd]));
    atomicAdd(&denom[row * NHEADS + hd], w);
    const float* hj = h + (size_t)col * NCOL + hd * OUTCH;
    float* op = out + (size_t)row * NCOL + hd * OUTCH;
    #pragma unroll
    for (int c = 0; c < OUTCH; ++c)
        atomicAdd(&op[c], hj[c] * w);
}

// ---- normalize + bias ----
__global__ void k_final(const float* __restrict__ denom, const float* __restrict__ bias,
                        float* __restrict__ out) {
    int t = blockIdx.x * blockDim.x + threadIdx.x;
    if (t >= NNODES * NCOL) return;
    int n = t >> 7, c = t & 127;
    out[t] = out[t] / (denom[n * NHEADS + (c >> 5)] + 1e-16f) + bias[c];
}

extern "C" void kernel_launch(void* const* d_in, const int* in_sizes, int n_in,
                              void* d_out, int out_size, void* d_ws, size_t ws_size,
                              hipStream_t stream) {
    const float* x    = (const float*)d_in[0];
    const int*   ei   = (const int*)d_in[1];
    const float* W    = (const float*)d_in[2];
    const float* att  = (const float*)d_in[3];
    const float* bias = (const float*)d_in[4];
    float* out = (float*)d_out;

    char* ws = (char*)d_ws;
    size_t off = 0;
    auto alloc = [&](size_t bytes) { char* p = ws + off; off = (off + bytes + 255) & ~(size_t)255; return p; };
    float*          h_buf = (float*)         alloc((size_t)NNODES * NCOL * 4);   // 25.6 MB
    unsigned short* Bst   = (unsigned short*)alloc((size_t)BELEMS * 2);
    float*          sl    = (float*)         alloc((size_t)NNODES * NHEADS * 4);
    float*          sr    = (float*)         alloc((size_t)NNODES * NHEADS * 4);
    unsigned int*   amax  = (unsigned int*)  alloc((size_t)NNODES * NHEADS * 4);
    float*          denom = (float*)         alloc((size_t)NNODES * NHEADS * 4);

    k_init   <<<(NNODES * NCOL + 255) / 256, 256, 0, stream>>>(out, amax, denom);
    k_stage_b<<<(NT * KB * 32 + 255) / 256, 256, 0, stream>>>(W, Bst);
    k_gemm   <<<(MT + 3) / 4, 128, 0, stream>>>(x, Bst, h_buf);          // 4 waves/block
    k_scores <<<(NNODES * NHEADS + 255) / 256, 256, 0, stream>>>(h_buf, att, sl, sr);
    k_edge_max<<<((NEDGES + NNODES) + 255) / 256, 256, 0, stream>>>(ei, sl, sr, amax);
    k_edge_agg<<<((NEDGES + NNODES) * NHEADS + 255) / 256, 256, 0, stream>>>(
        ei, sl, sr, amax, h_buf, denom, out);
    k_final  <<<(NNODES * NCOL + 255) / 256, 256, 0, stream>>>(denom, bias, out);
}
